// Block_62156766707749
// MI455X (gfx1250) — matrix operand order repaired and checked by reading, not compiled, not run on hardware
//
#include <hip/hip_runtime.h>

// ---------------------------------------------------------------------------
// MI455X (gfx1250) transformer block: LN -> QKV (WMMA) -> flash attention
// (WMMA) -> O-proj+residual -> LN -> FFN (WMMA, fused bias/relu/residual).
// Matrix math on v_wmma_f32_16x16x32_f16 (wave32, fp32 accumulate).
// GEMM stages tiles with GLOBAL_LOAD_ASYNC_TO_LDS_B128 (ASYNCcnt, inline asm)
// when s_wait_asynccnt is available, with a vmem+ds fallback otherwise.
// ---------------------------------------------------------------------------

typedef __attribute__((ext_vector_type(16))) _Float16 v16h;
typedef __attribute__((ext_vector_type(8)))  float    v8f;
typedef __attribute__((ext_vector_type(4)))  float    f4v;   // 16-byte vector load/store

union Frag16 {
    v16h     v;
    f4v      f[2];
    _Float16 h[16];
};

__device__ __forceinline__ v8f wmma16(const Frag16& a, const Frag16& b, v8f c) {
    // 8 args: (neg_a, A, neg_b, B, c_mod, C, reuse_a, reuse_b)
    return __builtin_amdgcn_wmma_f32_16x16x32_f16(false, a.v, false, b.v,
                                                  (short)0, c, false, false);
}

// ---- async global->LDS copy (CDNA5), with portable fallback ---------------
#if defined(__has_builtin)
#if __has_builtin(__builtin_amdgcn_s_wait_asynccnt)
#define USE_ASYNC_LDS 1
#endif
#endif
#ifndef USE_ASYNC_LDS
#define USE_ASYNC_LDS 0
#endif

__device__ __forceinline__ void lds_cp128(_Float16* l, const _Float16* g) {
#if USE_ASYNC_LDS
    // VDST = per-lane LDS byte offset (added to the LDS base),
    // VADDR = 64-bit global address, SADDR = off (NULL). Tracked by ASYNCcnt.
    unsigned loff = (unsigned)(unsigned long long)
        (__attribute__((address_space(3))) void*)l;
    asm volatile("global_load_async_to_lds_b128 %0, %1, off"
                 :: "v"(loff), "v"(g) : "memory");
#else
    *(f4v*)l = *(const f4v*)g;
#endif
}

__device__ __forceinline__ void async_fence() {
#if USE_ASYNC_LDS
    __builtin_amdgcn_s_wait_asynccnt(0);
#endif
}

// ---------------------------------------------------------------------------
// fp32 -> fp16 cast (weights), grid-stride
// ---------------------------------------------------------------------------
__global__ void cast_f16_kernel(const float* __restrict__ in,
                                _Float16* __restrict__ out, int n) {
    int i = blockIdx.x * blockDim.x + threadIdx.x;
    int stride = gridDim.x * blockDim.x;
    for (; i < n; i += stride) out[i] = (_Float16)in[i];
}

// ---------------------------------------------------------------------------
// LayerNorm over last dim (C=512) -> f16. One wave32 per row, 8 rows/block.
// ---------------------------------------------------------------------------
__global__ __launch_bounds__(256) void ln_f16_kernel(const float* __restrict__ x,
                                                     const float* __restrict__ g,
                                                     const float* __restrict__ b,
                                                     _Float16* __restrict__ out) {
    const int C = 512;
    int wave = threadIdx.x >> 5, lane = threadIdx.x & 31;
    size_t row = (size_t)blockIdx.x * 8 + wave;
    const float* xr = x + row * C;

    float xs[16];
    float s = 0.f;
#pragma unroll
    for (int i = 0; i < 16; i++) { xs[i] = xr[lane + 32 * i]; s += xs[i]; }
#pragma unroll
    for (int off = 16; off >= 1; off >>= 1) s += __shfl_xor(s, off, 32);
    float mu = s * (1.0f / 512.0f);

    float v = 0.f;
#pragma unroll
    for (int i = 0; i < 16; i++) { float d = xs[i] - mu; v += d * d; }
#pragma unroll
    for (int off = 16; off >= 1; off >>= 1) v += __shfl_xor(v, off, 32);
    float rs = rsqrtf(v * (1.0f / 512.0f) + 1e-5f);

    _Float16* orow = out + row * C;
#pragma unroll
    for (int i = 0; i < 16; i++) {
        int c = lane + 32 * i;
        orow[c] = (_Float16)((xs[i] - mu) * rs * g[c] + b[c]);
    }
}

// ---------------------------------------------------------------------------
// GEMM: out[M,N] = A[M,K](f16) @ Bw[N,K](f16)^T  (+bias, relu, +residual)
// mode 0: f32 out [M,N]; mode 1: f16 out [M,N];
// mode 2: f16 out scattered to V^T layout [b, h, d, T]  (row=(b,t), col=(h,d))
// Block tile 128x128, K-step 32, 8 waves: wave w -> M rows [16w,16w+16) x 128
// N cols => per K-step: 1 A-frag + 8 B-frags + 8 WMMAs (64 acc VGPRs).
// Double-buffered LDS; async global->LDS staging overlaps tile i+1 loads
// with tile i WMMAs.
// ---------------------------------------------------------------------------
#define SA 40   // LDS stride (halves), 80B: 16B aligned, bank-spread
#define SB 40
__global__ __launch_bounds__(256) void gemm_f16_kernel(
    const _Float16* __restrict__ A, const _Float16* __restrict__ Bw,
    int M, int N, int K,
    const float* __restrict__ bias, const float* __restrict__ resid,
    int relu, int mode, void* __restrict__ outp,
    int Tn, int Hn, int Dn) {
    __shared__ __align__(16) _Float16 lA[2][128 * SA];
    __shared__ __align__(16) _Float16 lB[2][128 * SB];

    int tid = threadIdx.x;
    int wave = tid >> 5, lane = tid & 31, lane15 = lane & 15, hi = lane >> 4;
    int nBase = blockIdx.x * 128, mBase = blockIdx.y * 128;

    v8f acc[8] = {};

    // staging map: 128 rows x 32 halves per tile; thread -> one 32B row chunk
    int sr = tid >> 1, sch = (tid & 1) * 16;
    const _Float16* gA = A + (size_t)(mBase + sr) * K + sch;
    const _Float16* gB = Bw + (size_t)(nBase + sr) * K + sch;

    int nk = K >> 5;
    // prologue: stage tile 0 into buffer 0
    {
        _Float16* la = &lA[0][sr * SA + sch];
        _Float16* lb = &lB[0][sr * SB + sch];
        lds_cp128(la, gA);     lds_cp128(la + 8, gA + 8);
        lds_cp128(lb, gB);     lds_cp128(lb + 8, gB + 8);
    }
    async_fence();
    __syncthreads();

    for (int i = 0; i < nk; ++i) {
        int cur = i & 1;
        if (i + 1 < nk) {  // stage next tile into the other buffer
            int k0 = (i + 1) << 5;
            _Float16* la = &lA[cur ^ 1][sr * SA + sch];
            _Float16* lb = &lB[cur ^ 1][sr * SB + sch];
            lds_cp128(la, gA + k0);     lds_cp128(la + 8, gA + k0 + 8);
            lds_cp128(lb, gB + k0);     lds_cp128(lb + 8, gB + k0 + 8);
        }

        // A frag (16x32, A-layout): lane m<16 -> row m, K 0..15;
        // lane m+16 -> row m, K 16..31
        Frag16 af;
        int ar = wave * 16 + lane15;
        af.f[0] = *(const f4v*)&lA[cur][ar * SA + hi * 16];
        af.f[1] = *(const f4v*)&lA[cur][ar * SA + hi * 16 + 8];

#pragma unroll
        for (int t = 0; t < 8; t++) {
            // B frag (32x16, B-layout): lane n<16 -> col n, K 0..15;
            // lane n+16 -> col n, K 16..31  (contiguous in [N,K] storage)
            Frag16 bf;
            int br = t * 16 + lane15;
            bf.f[0] = *(const f4v*)&lB[cur][br * SB + hi * 16];
            bf.f[1] = *(const f4v*)&lB[cur][br * SB + hi * 16 + 8];
            acc[t] = wmma16(af, bf, acc[t]);
        }
        async_fence();
        __syncthreads();
    }

    // Epilogue. C layout: VGPR r, lanes0-15 -> (M=r, N=lane); lanes16-31 -> (M=r+8)
#pragma unroll
    for (int t = 0; t < 8; t++) {
        int col = nBase + t * 16 + lane15;
        float bv = bias ? bias[col] : 0.0f;
#pragma unroll
        for (int r = 0; r < 8; r++) {
            int row = mBase + wave * 16 + r + hi * 8;
            float val = acc[t][r] + bv;
            if (relu) val = fmaxf(val, 0.0f);
            if (resid) val += resid[(size_t)row * N + col];
            if (mode == 0) {
                ((float*)outp)[(size_t)row * N + col] = val;
            } else if (mode == 1) {
                ((_Float16*)outp)[(size_t)row * N + col] = (_Float16)val;
            } else {
                int bb = row / Tn, tt = row % Tn, hh = col / Dn, dd = col % Dn;
                ((_Float16*)outp)[(((size_t)bb * Hn + hh) * Dn + dd) * Tn + tt] =
                    (_Float16)val;
            }
        }
    }
}

// ---------------------------------------------------------------------------
// Flash attention, causal. Grid: B*H*(T/64) blocks, 128 threads = 4 waves.
// Wave handles 16 query rows; streams 32-key tiles:
//   S(16x32) = Q(16x64) K^T     (2 accumulation chains of 2 WMMAs)
//   online softmax (half-wave shuffles match C-layout 16-lane row groups)
//   P -> f16 via per-wave LDS transpose (C-layout -> A-layout)
//   O(16x64) += P(16x32) V(32x64)   (4 WMMAs, V pre-transposed [b,h,d,T])
// ---------------------------------------------------------------------------
__global__ __launch_bounds__(128) void attn_kernel(const _Float16* __restrict__ q,
                                                   const _Float16* __restrict__ k,
                                                   const _Float16* __restrict__ vT,
                                                   _Float16* __restrict__ o) {
    const int Tn = 2048, Hn = 8, Dn = 64;
    __shared__ __align__(16) _Float16 pl[4][16 * 32];

    int wave = threadIdx.x >> 5, lane = threadIdx.x & 31;
    int lane15 = lane & 15, hi = lane >> 4;
    int qt = blockIdx.x & 31;          // T/64 = 32 query tiles
    int h  = (blockIdx.x >> 5) & 7;
    int bb = blockIdx.x >> 8;
    int qbase = qt * 64 + wave * 16;

    // Q frags (A-layout), head_dim split into two K=32 chunks
    Frag16 q0, q1;
    {
        const _Float16* qp = q + (((size_t)bb * Tn + qbase + lane15) * Hn + h) * Dn;
        q0.f[0] = *(const f4v*)(qp + hi * 16);
        q0.f[1] = *(const f4v*)(qp + hi * 16 + 8);
        q1.f[0] = *(const f4v*)(qp + 32 + hi * 16);
        q1.f[1] = *(const f4v*)(qp + 32 + hi * 16 + 8);
    }

    v8f oacc[4] = {};
    float mr[8], lr[8];
#pragma unroll
    for (int r = 0; r < 8; r++) { mr[r] = -1e30f; lr[r] = 0.f; }

    const float scale = 0.04419417382415922f;  // 512^-0.5 (ref scales by C)
    int jend = qbase + 16;                     // causal bound for this wave

    for (int j = 0; j < jend; j += 32) {
        // K frags (B-layout): y = key half (16 keys), c = head-dim chunk
        Frag16 kf[2][2];
#pragma unroll
        for (int y = 0; y < 2; y++) {
            const _Float16* kp =
                k + (((size_t)bb * Tn + j + y * 16 + lane15) * Hn + h) * Dn;
#pragma unroll
            for (int c = 0; c < 2; c++) {
                kf[y][c].f[0] = *(const f4v*)(kp + c * 32 + hi * 16);
                kf[y][c].f[1] = *(const f4v*)(kp + c * 32 + hi * 16 + 8);
            }
        }
        v8f s0 = {}, s1 = {};
        s0 = wmma16(q0, kf[0][0], s0); s0 = wmma16(q1, kf[0][1], s0);
        s1 = wmma16(q0, kf[1][0], s1); s1 = wmma16(q1, kf[1][1], s1);

        int key0 = j + lane15, key1 = key0 + 16;
#pragma unroll
        for (int r = 0; r < 8; r++) {
            int qr = qbase + r + hi * 8;
            float a = s0[r] * scale; if (key0 > qr) a = -1e30f;
            float c = s1[r] * scale; if (key1 > qr) c = -1e30f;
            float rowm = fmaxf(a, c);
#pragma unroll
            for (int off = 8; off >= 1; off >>= 1)
                rowm = fmaxf(rowm, __shfl_xor(rowm, off, 16));
            float mnew = fmaxf(mr[r], rowm);
            float al = __expf(mr[r] - mnew);
            float p0 = __expf(a - mnew);
            float p1 = __expf(c - mnew);
            float rs = p0 + p1;
#pragma unroll
            for (int off = 8; off >= 1; off >>= 1)
                rs += __shfl_xor(rs, off, 16);
            lr[r] = lr[r] * al + rs;
            mr[r] = mnew;
            oacc[0][r] *= al; oacc[1][r] *= al; oacc[2][r] *= al; oacc[3][r] *= al;
            // C-layout element (row = r+hi*8, col = lane15 [+16]) -> LDS
            int prow = r + hi * 8;
            pl[wave][prow * 32 + lane15]      = (_Float16)p0;
            pl[wave][prow * 32 + lane15 + 16] = (_Float16)p1;
        }
        // Re-read P in A-layout (same-wave DS ops are in-order)
        Frag16 pf;
        pf.f[0] = *(const f4v*)&pl[wave][lane15 * 32 + hi * 16];
        pf.f[1] = *(const f4v*)&pl[wave][lane15 * 32 + hi * 16 + 8];

#pragma unroll
        for (int t = 0; t < 4; t++) {
            // V frag (B-layout) from V^T [b,h,d,T]: lane n -> d=16t+n,
            // keys j..j+15 (lanes<16) / j+16..j+31 (lanes>=16), contiguous 32B
            Frag16 vf;
            const _Float16* vp =
                vT + (((size_t)bb * Hn + h) * Dn + t * 16 + lane15) * Tn + j + hi * 16;
            vf.f[0] = *(const f4v*)(vp);
            vf.f[1] = *(const f4v*)(vp + 8);
            oacc[t] = wmma16(pf, vf, oacc[t]);
        }
    }

    // Normalize and store f16 in natural [ (b,t), h*64+d ] layout
#pragma unroll
    for (int t = 0; t < 4; t++) {
#pragma unroll
        for (int r = 0; r < 8; r++) {
            int row = qbase + r + hi * 8;
            float val = oacc[t][r] / lr[r];
            o[(((size_t)bb * Tn + row) * Hn + h) * Dn + t * 16 + lane15] =
                (_Float16)val;
        }
    }
}

// ---------------------------------------------------------------------------
// Host-side orchestration
// ---------------------------------------------------------------------------
extern "C" void kernel_launch(void* const* d_in, const int* in_sizes, int n_in,
                              void* d_out, int out_size, void* d_ws, size_t ws_size,
                              hipStream_t stream) {
    (void)in_sizes; (void)n_in; (void)out_size; (void)ws_size;
    const int Bn = 4, Tn = 2048, C = 512, F = 2048, Hn = 8, Dn = 64;
    const int M = Bn * Tn;  // 8192

    const float* x    = (const float*)d_in[0];
    const float* Wq   = (const float*)d_in[1];
    const float* Wk   = (const float*)d_in[2];
    const float* Wv   = (const float*)d_in[3];
    const float* Wo   = (const float*)d_in[4];
    const float* bo   = (const float*)d_in[5];
    const float* g1   = (const float*)d_in[6];
    const float* b1   = (const float*)d_in[7];
    const float* g2   = (const float*)d_in[8];
    const float* b2   = (const float*)d_in[9];
    const float* Wff1 = (const float*)d_in[10];
    const float* bff1 = (const float*)d_in[11];
    const float* Wff2 = (const float*)d_in[12];
    const float* bff2 = (const float*)d_in[13];

    char* ws = (char*)d_ws;
    size_t off = 0;
    auto alloc = [&](size_t bytes) -> void* {
        void* p = ws + off;
        off += (bytes + 255) & ~(size_t)255;
        return p;
    };
    _Float16* Wq_h   = (_Float16*)alloc((size_t)C * C * 2);
    _Float16* Wk_h   = (_Float16*)alloc((size_t)C * C * 2);
    _Float16* Wv_h   = (_Float16*)alloc((size_t)C * C * 2);
    _Float16* Wo_h   = (_Float16*)alloc((size_t)C * C * 2);
    _Float16* Wff1_h = (_Float16*)alloc((size_t)F * C * 2);
    _Float16* Wff2_h = (_Float16*)alloc((size_t)C * F * 2);
    _Float16* h1     = (_Float16*)alloc((size_t)M * C * 2);
    _Float16* qh     = (_Float16*)alloc((size_t)M * C * 2);
    _Float16* kh     = (_Float16*)alloc((size_t)M * C * 2);
    _Float16* vTh    = (_Float16*)alloc((size_t)M * C * 2);
    _Float16* attnh  = (_Float16*)alloc((size_t)M * C * 2);
    float*    x1     = (float*)   alloc((size_t)M * C * 4);
    _Float16* h2     = (_Float16*)alloc((size_t)M * C * 2);
    _Float16* ff1    = (_Float16*)alloc((size_t)M * F * 2);

    // 1) weights -> f16
    cast_f16_kernel<<<256, 256, 0, stream>>>(Wq, Wq_h, C * C);
    cast_f16_kernel<<<256, 256, 0, stream>>>(Wk, Wk_h, C * C);
    cast_f16_kernel<<<256, 256, 0, stream>>>(Wv, Wv_h, C * C);
    cast_f16_kernel<<<256, 256, 0, stream>>>(Wo, Wo_h, C * C);
    cast_f16_kernel<<<512, 256, 0, stream>>>(Wff1, Wff1_h, F * C);
    cast_f16_kernel<<<512, 256, 0, stream>>>(Wff2, Wff2_h, C * F);

    // 2) h1 = LN(x, g1, b1)
    ln_f16_kernel<<<M / 8, 256, 0, stream>>>(x, g1, b1, h1);

    // 3) Q, K, V projections (V stored transposed [b,h,d,T])
    dim3 gqkv(C / 128, M / 128);
    gemm_f16_kernel<<<gqkv, 256, 0, stream>>>(h1, Wq_h, M, C, C, nullptr, nullptr,
                                              0, 1, qh, Tn, Hn, Dn);
    gemm_f16_kernel<<<gqkv, 256, 0, stream>>>(h1, Wk_h, M, C, C, nullptr, nullptr,
                                              0, 1, kh, Tn, Hn, Dn);
    gemm_f16_kernel<<<gqkv, 256, 0, stream>>>(h1, Wv_h, M, C, C, nullptr, nullptr,
                                              0, 2, vTh, Tn, Hn, Dn);

    // 4) causal flash attention
    attn_kernel<<<Bn * Hn * (Tn / 64), 128, 0, stream>>>(qh, kh, vTh, attnh);

    // 5) x1 = x + attn @ Wo^T + bo
    gemm_f16_kernel<<<gqkv, 256, 0, stream>>>(attnh, Wo_h, M, C, C, bo, x,
                                              0, 0, x1, Tn, Hn, Dn);

    // 6) h2 = LN(x1, g2, b2)
    ln_f16_kernel<<<M / 8, 256, 0, stream>>>(x1, g2, b2, h2);

    // 7) ff1 = relu(h2 @ Wff1^T + bff1)
    dim3 gff1(F / 128, M / 128);
    gemm_f16_kernel<<<gff1, 256, 0, stream>>>(h2, Wff1_h, M, F, C, bff1, nullptr,
                                              1, 1, ff1, Tn, Hn, Dn);

    // 8) out = x1 + ff1 @ Wff2^T + bff2
    dim3 gff2(C / 128, M / 128);
    gemm_f16_kernel<<<gff2, 256, 0, stream>>>(ff1, Wff2_h, M, C, F, bff2, x1,
                                              0, 0, (float*)d_out, Tn, Hn, Dn);
}
